// Attention_23682449670526
// MI455X (gfx1250) — matrix operand order repaired and checked by reading, not compiled, not run on hardware
//
#include <hip/hip_runtime.h>
#include <hip/hip_bf16.h>

typedef __attribute__((ext_vector_type(16))) _Float16 v16h;
typedef __attribute__((ext_vector_type(8)))  float    v8f;

#define DIM   768
#define HEADS 12
#define HD    64
#define BATCH 4
#define SEQ   2048
#define SROW  2052                   // LDS row stride (floats); 2052 % 64 = 4 -> conflict-free rows

__device__ __forceinline__ v8f wmma_f16(v16h a, v16h b, v8f c) {
    // D = A(16x32 f16) * B(32x16 f16) + C(16x16 f32)
    return __builtin_amdgcn_wmma_f32_16x16x32_f16(
        /*neg_a=*/false, a, /*neg_b=*/false, b,
        /*c_mod=*/(short)0, c, /*reuse_a=*/false, /*reuse_b=*/false);
}

// ---- fragment loaders (ISA 7.12.2 layouts, wave32) --------------------------
// A 16x32 f16: lane l: row = l&15; halves K = hi*8+i and K = 16+hi*8+i (two 16B runs)
__device__ __forceinline__ v16h load_a_f16(const _Float16* __restrict__ src, int lda,
                                           int m0, int k0, int lane) {
    int l = lane & 15, hi = lane >> 4;
    const _Float16* p = src + (size_t)(m0 + l) * lda + k0 + hi * 8;
    v16h a;
#pragma unroll
    for (int i = 0; i < 8; ++i) { a[i] = p[i]; a[8 + i] = p[16 + i]; }
    return a;
}

// B 32x16 f16 from transposed storage srcT[n][k] (row n contiguous in k):
// lane = column n0+(l&15); K = hi*8+i and K = 16+hi*8+i (two 16B runs)
__device__ __forceinline__ v16h load_bT_f16(const _Float16* __restrict__ srcT, int ld,
                                            int n0, int k0, int lane) {
    int l = lane & 15, hi = lane >> 4;
    const _Float16* p = srcT + (size_t)(n0 + l) * ld + k0 + hi * 8;
    v16h b;
#pragma unroll
    for (int i = 0; i < 8; ++i) { b[i] = p[i]; b[8 + i] = p[16 + i]; }
    return b;
}

// A fragment from f32 LDS score/probability strip (rows = 16 queries)
__device__ __forceinline__ v16h load_a_lds(const float* S, int k0, int lane) {
    int l = lane & 15, hi = lane >> 4;
    const float* p = S + (size_t)l * SROW + k0 + hi * 8;
    v16h a;
#pragma unroll
    for (int i = 0; i < 8; ++i) { a[i] = (_Float16)p[i]; a[8 + i] = (_Float16)p[16 + i]; }
    return a;
}

// ============================================================================
// Precondition kernels (one-shot, tiny vs. GEMM cost)
// ============================================================================
__global__ __launch_bounds__(256) void cvt_kernel(const float* __restrict__ src,
                                                  _Float16* __restrict__ dst) {
    const size_t i = ((size_t)blockIdx.x * 256 + threadIdx.x) * 4;
#pragma unroll
    for (int j = 0; j < 4; ++j) dst[i + j] = (_Float16)src[i + j];
}

// src[rows][cols] f32 -> dst[cols][rows] f16 (coalesced writes; reads served by 192MB L2)
__global__ __launch_bounds__(256) void transpose_cvt_kernel(const float* __restrict__ src,
                                                            _Float16* __restrict__ dst,
                                                            int rows, int cols) {
    const size_t idx = (size_t)blockIdx.x * 256 + threadIdx.x;   // over dst elements
    const int r = (int)(idx / rows);                              // = src column
    const int c = (int)(idx % rows);                              // = src row
    dst[idx] = (_Float16)src[(size_t)c * cols + r];
}

// ============================================================================
// Kernel 1: QKV = Xh @ Wqkv ; scatter into Q[B,H,N,64], K[B,H,N,64], Vt[B,H,64,N]
// wave tile = 32x32 (4 WMMAs / 8 b128 loads per k-step) ; 256 x 72 wave tiles
// ============================================================================
__device__ __forceinline__ void scatter_qkv(const v8f& acc, int m0, int n0, int lane,
                                            _Float16* __restrict__ Qo,
                                            _Float16* __restrict__ Ko,
                                            _Float16* __restrict__ Vt) {
    const int l = lane & 15, rbase = (lane >> 4) ? 8 : 0;
    const int n  = n0 + l;
    const int sel = n / DIM, nn = n % DIM;
    const int h = nn >> 6, d = nn & 63;
#pragma unroll
    for (int j = 0; j < 8; ++j) {
        const int m  = m0 + rbase + j;
        const int bb = m >> 11, pos = m & (SEQ - 1);
        const _Float16 v = (_Float16)acc[j];
        const size_t bh = (size_t)(bb * HEADS + h);
        if (sel == 0)      Qo[(bh * SEQ + pos) * HD + d] = v;
        else if (sel == 1) Ko[(bh * SEQ + pos) * HD + d] = v;
        else               Vt[(bh * HD + d) * SEQ + pos] = v;
    }
}

__global__ __launch_bounds__(128) void qkv_gemm_kernel(
    const _Float16* __restrict__ Xh, const _Float16* __restrict__ Wt,   // Wt: [2304][768]
    _Float16* __restrict__ Qo, _Float16* __restrict__ Ko, _Float16* __restrict__ Vt) {
    const int lane = threadIdx.x & 31;
    const int wave = threadIdx.x >> 5;
    const int wv   = blockIdx.x * 4 + wave;          // 0 .. 18431
    const int mt   = wv / 72, nt = wv % 72;          // 256 x 72 tiles (M,N step 32)
    const int m0   = mt * 32, n0 = nt * 32;

    v8f acc00 = {}, acc01 = {}, acc10 = {}, acc11 = {};
#pragma unroll 2
    for (int k0 = 0; k0 < DIM; k0 += 32) {
        v16h a0 = load_a_f16(Xh, DIM, m0,      k0, lane);
        v16h a1 = load_a_f16(Xh, DIM, m0 + 16, k0, lane);
        v16h b0 = load_bT_f16(Wt, DIM, n0,      k0, lane);
        v16h b1 = load_bT_f16(Wt, DIM, n0 + 16, k0, lane);
        acc00 = wmma_f16(a0, b0, acc00);
        acc01 = wmma_f16(a0, b1, acc01);
        acc10 = wmma_f16(a1, b0, acc10);
        acc11 = wmma_f16(a1, b1, acc11);
    }
    scatter_qkv(acc00, m0,      n0,      lane, Qo, Ko, Vt);
    scatter_qkv(acc01, m0,      n0 + 16, lane, Qo, Ko, Vt);
    scatter_qkv(acc10, m0 + 16, n0,      lane, Qo, Ko, Vt);
    scatter_qkv(acc11, m0 + 16, n0 + 16, lane, Qo, Ko, Vt);
}

// ============================================================================
// Kernel 2: softmax(Q Kt * scale) V per (b, h, 16-query tile)
// grid: 6144 blocks x 128 thr ; dynamic LDS: 16*SROW floats (~128 KB < 320 KB WGP)
// ============================================================================
__global__ __launch_bounds__(128) void attn_kernel(
    const _Float16* __restrict__ Qi, const _Float16* __restrict__ Ki,
    const _Float16* __restrict__ Vt, _Float16* __restrict__ AO) {
    extern __shared__ float S[];                      // [16][SROW]
    __shared__ float pbuf[16][8];
    __shared__ float rowsum[16];

    const int tid  = threadIdx.x;
    const int lane = tid & 31;
    const int wave = tid >> 5;
    const int qt   = blockIdx.x & 127;                // 128 q-tiles
    const int bh   = blockIdx.x >> 7;                 // 0..47
    const int bb   = bh / HEADS, h = bh % HEADS;
    const int q0   = qt * 16;

    const _Float16* Qbh = Qi + (size_t)bh * SEQ * HD;
    const _Float16* Kbh = Ki + (size_t)bh * SEQ * HD;
    const _Float16* Vbh = Vt + (size_t)bh * HD * SEQ;

    // ---- phase 1: S = scale * Q Kt  (each wave does every 4th 16-key tile) ----
    const v16h aq0 = load_a_f16(Qbh, HD, q0, 0,  lane);
    const v16h aq1 = load_a_f16(Qbh, HD, q0, 32, lane);
    const int l = lane & 15, rbase = (lane >> 4) ? 8 : 0;
    for (int kt = wave; kt < 128; kt += 4) {
        const int n0 = kt * 16;
        v8f acc = {};
        v16h b0 = load_bT_f16(Kbh, HD, n0, 0,  lane);   // K rows contiguous in d
        v16h b1 = load_bT_f16(Kbh, HD, n0, 32, lane);
        acc = wmma_f16(aq0, b0, acc);
        acc = wmma_f16(aq1, b1, acc);
        const int col = n0 + l;
#pragma unroll
        for (int j = 0; j < 8; ++j)
            S[(size_t)(rbase + j) * SROW + col] = acc[j] * 0.125f;  // 1/sqrt(64)
    }
    __syncthreads();

    // ---- phase 2: row softmax over LDS (8 threads per row) --------------------
    {
        const int row = tid >> 3, t = tid & 7;
        float m = -3.0e38f;
        for (int k = t; k < SEQ; k += 8) m = fmaxf(m, S[(size_t)row * SROW + k]);
        pbuf[row][t] = m;
        __syncthreads();
        m = pbuf[row][0];
#pragma unroll
        for (int i = 1; i < 8; ++i) m = fmaxf(m, pbuf[row][i]);
        float s = 0.0f;
        for (int k = t; k < SEQ; k += 8) {
            float e = __expf(S[(size_t)row * SROW + k] - m);
            S[(size_t)row * SROW + k] = e;
            s += e;
        }
        __syncthreads();
        pbuf[row][t] = s;
        __syncthreads();
        if (t == 0) {
            float ss = 0.0f;
#pragma unroll
            for (int i = 0; i < 8; ++i) ss += pbuf[row][i];
            rowsum[row] = ss;
        }
        __syncthreads();
    }

    // ---- phase 3: O = P @ V ; each wave owns one 16-wide slice of hd ----------
    {
        const int n0 = wave * 16;                     // output d-columns
        v8f acc = {};
        for (int kb = 0; kb < SEQ; kb += 32) {
            v16h a  = load_a_lds(S, kb, lane);
            v16h bv = load_bT_f16(Vbh, SEQ, n0, kb, lane); // Vt rows contiguous in key
            acc = wmma_f16(a, bv, acc);
        }
        const int c = h * HD + n0 + l;                // column in [0,768)
#pragma unroll
        for (int j = 0; j < 8; ++j) {
            const int row = rbase + j;
            const float v = acc[j] / rowsum[row];
            AO[((size_t)bb * SEQ + q0 + row) * DIM + c] = (_Float16)v;
        }
    }
}

// ============================================================================
// Kernel 3: out = AO @ W_proj + b_proj   (f32 out)
// wave tile = 32x32 ; 256 x 24 wave tiles ; 4 waves/block
// ============================================================================
__global__ __launch_bounds__(128) void proj_gemm_kernel(
    const _Float16* __restrict__ AO, const _Float16* __restrict__ Wt,   // Wt: [768][768]
    const float* __restrict__ b_proj, float* __restrict__ out) {
    const int lane = threadIdx.x & 31;
    const int wave = threadIdx.x >> 5;
    const int wv   = blockIdx.x * 4 + wave;
    const int mt   = wv / 24, nt = wv % 24;          // 256 x 24 tiles (M,N step 32)
    const int m0   = mt * 32, n0 = nt * 32;

    v8f acc00 = {}, acc01 = {}, acc10 = {}, acc11 = {};
#pragma unroll 2
    for (int k0 = 0; k0 < DIM; k0 += 32) {
        v16h a0 = load_a_f16(AO, DIM, m0,      k0, lane);
        v16h a1 = load_a_f16(AO, DIM, m0 + 16, k0, lane);
        v16h b0 = load_bT_f16(Wt, DIM, n0,      k0, lane);
        v16h b1 = load_bT_f16(Wt, DIM, n0 + 16, k0, lane);
        acc00 = wmma_f16(a0, b0, acc00);
        acc01 = wmma_f16(a0, b1, acc01);
        acc10 = wmma_f16(a1, b0, acc10);
        acc11 = wmma_f16(a1, b1, acc11);
    }

    const int l = lane & 15, rbase = (lane >> 4) ? 8 : 0;
#pragma unroll
    for (int t = 0; t < 4; ++t) {
        const v8f& acc = (t == 0) ? acc00 : (t == 1) ? acc01 : (t == 2) ? acc10 : acc11;
        const int mbase = m0 + ((t >> 1) ? 16 : 0);
        const int n = n0 + ((t & 1) ? 16 : 0) + l;
        const float bias = b_proj[n];
#pragma unroll
        for (int j = 0; j < 8; ++j)
            out[(size_t)(mbase + rbase + j) * DIM + n] = acc[j] + bias;
    }
}

// ============================================================================
extern "C" void kernel_launch(void* const* d_in, const int* in_sizes, int n_in,
                              void* d_out, int out_size, void* d_ws, size_t ws_size,
                              hipStream_t stream) {
    const float* x      = (const float*)d_in[0];
    const float* w_qkv  = (const float*)d_in[1];
    const float* w_proj = (const float*)d_in[2];
    const float* b_proj = (const float*)d_in[3];
    float* out = (float*)d_out;

    const size_t TSZ = (size_t)BATCH * HEADS * SEQ * HD;   // 6,291,456 elems
    _Float16* Q    = (_Float16*)d_ws;
    _Float16* K    = Q    + TSZ;
    _Float16* Vt   = K    + TSZ;
    _Float16* AO   = Vt   + TSZ;                           // [B,N,768] f16
    _Float16* Xh   = AO   + TSZ;                           // [8192,768] f16 (same count)
    _Float16* Wqt  = Xh   + TSZ;                           // [2304,768] f16
    _Float16* Wpt  = Wqt  + (size_t)(3 * DIM) * DIM;       // [768,768]  f16
    // total ws: ~67.6 MB

    // 0) precondition: f16 convert + weight transposes
    cvt_kernel<<<dim3(6144), dim3(256), 0, stream>>>(x, Xh);
    transpose_cvt_kernel<<<dim3(6912), dim3(256), 0, stream>>>(w_qkv, Wqt, DIM, 3 * DIM);
    transpose_cvt_kernel<<<dim3(2304), dim3(256), 0, stream>>>(w_proj, Wpt, DIM, DIM);

    // 1) QKV GEMM: 256*72 wave tiles (32x32), 4 waves/block
    qkv_gemm_kernel<<<dim3(4608), dim3(128), 0, stream>>>(Xh, Wqt, Q, K, Vt);

    // 2) attention: one block per (b,h,q-tile); LDS strip 16*SROW floats
    const size_t smem = (size_t)16 * SROW * sizeof(float); // 131,328 B
    attn_kernel<<<dim3(BATCH * HEADS * (SEQ / 16)), dim3(128), smem, stream>>>(Q, K, Vt, AO);

    // 3) projection GEMM: 256*24 wave tiles (32x32), 4 waves/block
    proj_gemm_kernel<<<dim3(1536), dim3(128), 0, stream>>>(AO, Wpt, b_proj, out);
}
